// RGCNEncoder_44856638439570
// MI455X (gfx1250) — matrix operand order repaired and checked by reading, not compiled, not run on hardware
//
#include <hip/hip_runtime.h>

// ---------------------------------------------------------------------------
// RGCN encoder (2 layers) for MI455X / gfx1250, wave32 + WMMA f16->f32.
//
// Strategy:
//   * Fold bases into per-relation W_r (f16, transposed [n][k]) once.
//   * Stage node features in f16 once per layer (halves the edge-gather
//     traffic, removes all cvt ops from the WMMA hot loop).
//   * Counting-sort edges by relation, buckets 16-aligned, so each 16-edge
//     tile has a single W_r and can use v_wmma_f32_16x16x32_f16.
//   * h = x @ root + bias (dense WMMA), atomic f32 scatter of inv_deg-scaled
//     messages, fused ReLU+f16-convert between layers.
// ---------------------------------------------------------------------------

#define NN    50000
#define NE    800000
#define NREL  8
#define NB    4
#define D     64
#define EPAD  (NE + NREL * 16)   // relation buckets padded to 16

typedef __attribute__((ext_vector_type(16))) _Float16 v16h;
typedef __attribute__((ext_vector_type(8)))  _Float16 v8h;
typedef __attribute__((ext_vector_type(8)))  float    v8f;

__device__ __forceinline__ v8f wmma_step(v16h a, v16h b, v8f c) {
  // D = A(16x32 f16) * B(32x16 f16) + C(16x16 f32)
  return __builtin_amdgcn_wmma_f32_16x16x32_f16(
      /*neg_a=*/false, a, /*neg_b=*/false, b,
      /*c_mod=*/(short)0, c, /*reuse_a=*/false, /*reuse_b=*/false);
}

// A fragment from an f16 row: lane covers row M = lane&15, half h = lane>>4.
// Element idx 0..15 -> K = kBase + 16h + idx
__device__ __forceinline__ v16h load_a_frag(const _Float16* __restrict__ row,
                                            int kBase, int h) {
  v8h lo = *(const v8h*)(row + kBase + 16 * h);       // 16B load
  v8h hi = *(const v8h*)(row + kBase + 8 + 16 * h);   // 16B load
  v16h a;
#pragma unroll
  for (int i = 0; i < 8; ++i) { a[i] = lo[i]; a[8 + i] = hi[i]; }
  return a;
}

// ------------------------- setup kernels -----------------------------------

__global__ void ws_init_kernel(int* __restrict__ deg, int* __restrict__ cc,
                               int* __restrict__ sortedE) {
  int t = blockIdx.x * blockDim.x + threadIdx.x;
  if (t < NREL * NN) deg[t] = 0;
  if (t < 16) cc[t] = 0;          // counts[0..7], cursors[8..15]
  if (t < EPAD) sortedE[t] = -1;
}

__global__ void deg_count_kernel(const int* __restrict__ ei,
                                 const int* __restrict__ et,
                                 int* __restrict__ deg,
                                 int* __restrict__ counts) {
  __shared__ int cnt[NREL];
  if (threadIdx.x < NREL) cnt[threadIdx.x] = 0;
  __syncthreads();
  int e = blockIdx.x * blockDim.x + threadIdx.x;
  if (e < NE) {
    int r = et[e];
    int d = ei[NE + e];
    atomicAdd(&deg[r * NN + d], 1);
    atomicAdd(&cnt[r], 1);
  }
  __syncthreads();
  if (threadIdx.x < NREL) atomicAdd(&counts[threadIdx.x], cnt[threadIdx.x]);
}

__global__ void make_offsets_kernel(const int* __restrict__ counts,
                                    int* __restrict__ bstart) {
  if (blockIdx.x == 0 && threadIdx.x == 0) {
    int acc = 0;
    for (int r = 0; r < NREL; ++r) {
      bstart[r] = acc;
      acc += (counts[r] + 15) & ~15;     // keep every bucket 16-aligned
    }
    bstart[NREL] = acc;                   // padded edge count EP
  }
}

__global__ void inv_deg_kernel(const int* __restrict__ ei,
                               const int* __restrict__ et,
                               const int* __restrict__ deg,
                               float* __restrict__ inv) {
  int e = blockIdx.x * blockDim.x + threadIdx.x;
  if (e < NE) {
    int r = et[e];
    int d = ei[NE + e];
    int dg = deg[r * NN + d];
    inv[e] = 1.0f / (float)(dg < 1 ? 1 : dg);
  }
}

__global__ void bucket_scatter_kernel(const int* __restrict__ et,
                                      const int* __restrict__ bstart,
                                      int* __restrict__ cursors,
                                      int* __restrict__ sortedE) {
  __shared__ int cnt[NREL];
  __shared__ int baseOff[NREL];
  if (threadIdx.x < NREL) cnt[threadIdx.x] = 0;
  __syncthreads();
  int e = blockIdx.x * blockDim.x + threadIdx.x;
  int r = 0, my = 0;
  bool valid = (e < NE);
  if (valid) {
    r = et[e];
    my = atomicAdd(&cnt[r], 1);
  }
  __syncthreads();
  if (threadIdx.x < NREL)
    baseOff[threadIdx.x] = atomicAdd(&cursors[threadIdx.x], cnt[threadIdx.x]);
  __syncthreads();
  if (valid) sortedE[bstart[r] + baseOff[r] + my] = e;
}

// W_r = sum_b comp[r,b] * basis[b]; stored transposed [r][n][k] in f16.
// Also rootT[n][k] = root[k][n] in f16.
__global__ void prep_weights_kernel(const float* __restrict__ basis,
                                    const float* __restrict__ comp,
                                    const float* __restrict__ root,
                                    _Float16* __restrict__ Wt,
                                    _Float16* __restrict__ rootT) {
  int t = blockIdx.x * blockDim.x + threadIdx.x;
  const int WTOT = NREL * D * D;
  if (t < WTOT) {
    int r = t >> 12;
    int rem = t & 4095;
    int n = rem >> 6;
    int k = rem & 63;
    float s = 0.0f;
#pragma unroll
    for (int b = 0; b < NB; ++b)
      s += comp[r * NB + b] * basis[b * D * D + k * D + n];
    Wt[r * D * D + n * D + k] = (_Float16)s;
  } else if (t < WTOT + D * D) {
    int u = t - WTOT;
    int n = u >> 6;
    int k = u & 63;
    rootT[n * D + k] = (_Float16)root[k * D + n];
  }
}

// f32 -> f16 staging of node features.
__global__ void cvt_f16_kernel(const float* __restrict__ src,
                               _Float16* __restrict__ dst, int n) {
  int t = blockIdx.x * blockDim.x + threadIdx.x;
  if (t < n) dst[t] = (_Float16)src[t];
}

// Fused ReLU + f32 -> f16 staging (between layer 1 and layer 2).
__global__ void relu_cvt_kernel(const float* __restrict__ src,
                                _Float16* __restrict__ dst, int n) {
  int t = blockIdx.x * blockDim.x + threadIdx.x;
  if (t < n) dst[t] = (_Float16)fmaxf(src[t], 0.0f);
}

// ------------------------- dense root GEMM ---------------------------------
// Out[i][:] = X16[i][:] @ W (W stored transposed [n][k] f16) + bias[:]
__global__ void dense_rowgemm_kernel(const _Float16* __restrict__ X16,
                                     const _Float16* __restrict__ Wt,
                                     const float* __restrict__ bias,
                                     float* __restrict__ Out, int nrows) {
  int wavesPerBlk = blockDim.x >> 5;
  int tile = blockIdx.x * wavesPerBlk + (threadIdx.x >> 5);
  int lane = threadIdx.x & 31;
  int rowBase = tile * 16;
  if (rowBase >= nrows) return;                   // uniform per wave
  int h = lane >> 4;
  int n = lane & 15;
  const _Float16* xrow = X16 + (size_t)(rowBase + n) * D;
  v16h a0 = load_a_frag(xrow, 0, h);
  v16h a1 = load_a_frag(xrow, 32, h);
#pragma unroll
  for (int c = 0; c < 4; ++c) {
    int col = c * 16 + n;
    v16h b0 = *(const v16h*)(Wt + col * D + 16 * h);
    v16h b1 = *(const v16h*)(Wt + col * D + 32 + 16 * h);
    v8f acc = {};
    acc = wmma_step(a0, b0, acc);
    acc = wmma_step(a1, b1, acc);
    float bv = bias[col];
#pragma unroll
    for (int q = 0; q < 8; ++q) {
      int m = rowBase + q + 8 * h;
      Out[(size_t)m * D + col] = acc[q] + bv;
    }
  }
}

// ------------------------- edge message GEMM -------------------------------
// For each 16-edge tile (single relation): M = X16[src] @ W_r, scaled by
// inv_deg, atomically accumulated into Out[dst].
__global__ void msg_gemm_kernel(const _Float16* __restrict__ X16,
                                const _Float16* __restrict__ Wt,
                                const int* __restrict__ sortedE,
                                const int* __restrict__ bstart,
                                const int* __restrict__ ei,
                                const float* __restrict__ inv,
                                float* __restrict__ Out) {
  __shared__ int   sE[8][16];
  __shared__ int   sD[8][16];
  __shared__ float sI[8][16];
  int w = threadIdx.x >> 5;
  int lane = threadIdx.x & 31;
  int tile = blockIdx.x * (blockDim.x >> 5) + w;
  int base = tile * 16;
  int EP = bstart[NREL];
  bool active = (base < EP);
  if (active && lane < 16) {
    int e = sortedE[base + lane];
    sE[w][lane] = e;
    if (e >= 0) {
      sD[w][lane] = ei[NE + e];
      sI[w][lane] = inv[e];
    } else {
      sD[w][lane] = 0;
      sI[w][lane] = 0.0f;
    }
  }
  __syncthreads();
  if (!active) return;                            // uniform per wave

  // Which relation bucket is this tile in? (buckets are 16-aligned)
  int r = 0;
#pragma unroll
  for (int i = 1; i < NREL; ++i)
    if (base >= bstart[i]) r = i;
  const _Float16* W = Wt + (size_t)r * D * D;

  int h = lane >> 4;
  int n = lane & 15;
  int e = sE[w][n];
  v16h a0 = {}, a1 = {};
  if (e >= 0) {
    int src = ei[e];
    const _Float16* xrow = X16 + (size_t)src * D;
    a0 = load_a_frag(xrow, 0, h);
    a1 = load_a_frag(xrow, 32, h);
  }
#pragma unroll
  for (int c = 0; c < 4; ++c) {
    int col = c * 16 + n;
    v16h b0 = *(const v16h*)(W + col * D + 16 * h);
    v16h b1 = *(const v16h*)(W + col * D + 32 + 16 * h);
    v8f acc = {};
    acc = wmma_step(a0, b0, acc);
    acc = wmma_step(a1, b1, acc);
#pragma unroll
    for (int q = 0; q < 8; ++q) {
      int m = q + 8 * h;                          // row within tile
      int ee = sE[w][m];
      if (ee >= 0) {
        float v = acc[q] * sI[w][m];
        atomicAdd(&Out[(size_t)sD[w][m] * D + col], v);
      }
    }
  }
}

// ---------------------------------------------------------------------------

extern "C" void kernel_launch(void* const* d_in, const int* in_sizes, int n_in,
                              void* d_out, int out_size, void* d_ws,
                              size_t ws_size, hipStream_t stream) {
  (void)in_sizes; (void)n_in; (void)out_size; (void)ws_size;

  const float* x      = (const float*)d_in[0];
  const int*   ei     = (const int*)d_in[1];   // [2][NE]: row0=src, row1=dst
  const int*   et     = (const int*)d_in[2];   // [NE]
  const float* basis1 = (const float*)d_in[3];
  const float* comp1  = (const float*)d_in[4];
  const float* root1  = (const float*)d_in[5];
  const float* bias1  = (const float*)d_in[6];
  const float* basis2 = (const float*)d_in[7];
  const float* comp2  = (const float*)d_in[8];
  const float* root2  = (const float*)d_in[9];
  const float* bias2  = (const float*)d_in[10];
  float* out = (float*)d_out;

  // workspace carve-up (256B aligned)
  char* p = (char*)d_ws;
  auto carve = [&](size_t bytes) -> void* {
    void* r = (void*)p;
    p += (bytes + 255) & ~(size_t)255;
    return r;
  };
  int*      deg     = (int*)carve((size_t)NREL * NN * 4);
  int*      cc      = (int*)carve(16 * 4);          // counts[0..7] cursors[8..15]
  int*      bstart  = (int*)carve(16 * 4);          // [0..8]
  int*      sortedE = (int*)carve((size_t)EPAD * 4);
  float*    inv     = (float*)carve((size_t)NE * 4);
  _Float16* Wt1     = (_Float16*)carve((size_t)NREL * D * D * 2);
  _Float16* rT1     = (_Float16*)carve((size_t)D * D * 2);
  _Float16* Wt2     = (_Float16*)carve((size_t)NREL * D * D * 2);
  _Float16* rT2     = (_Float16*)carve((size_t)D * D * 2);
  float*    hbuf    = (float*)carve((size_t)NN * D * 4);
  _Float16* x16     = (_Float16*)carve((size_t)NN * D * 2);
  _Float16* h16     = (_Float16*)carve((size_t)NN * D * 2);

  const int TB = 256;
  // --- graph structure prep ---
  ws_init_kernel<<<(EPAD + TB - 1) / TB, TB, 0, stream>>>(deg, cc, sortedE);
  deg_count_kernel<<<(NE + TB - 1) / TB, TB, 0, stream>>>(ei, et, deg, cc);
  make_offsets_kernel<<<1, 32, 0, stream>>>(cc, bstart);
  inv_deg_kernel<<<(NE + TB - 1) / TB, TB, 0, stream>>>(ei, et, deg, inv);
  bucket_scatter_kernel<<<(NE + TB - 1) / TB, TB, 0, stream>>>(et, bstart,
                                                               cc + 8, sortedE);
  // --- weight prep (both layers) ---
  const int PREP = (NREL * D * D + D * D + TB - 1) / TB;
  prep_weights_kernel<<<PREP, TB, 0, stream>>>(basis1, comp1, root1, Wt1, rT1);
  prep_weights_kernel<<<PREP, TB, 0, stream>>>(basis2, comp2, root2, Wt2, rT2);

  const int NELT        = NN * D;
  const int denseBlocks = (NN / 16 + 7) / 8;        // 8 waves/block, 16 rows/wave
  const int msgBlocks   = (EPAD / 16 + 7) / 8;

  // --- layer 1 ---
  cvt_f16_kernel<<<(NELT + TB - 1) / TB, TB, 0, stream>>>(x, x16, NELT);
  dense_rowgemm_kernel<<<denseBlocks, TB, 0, stream>>>(x16, rT1, bias1, hbuf,
                                                       NN);
  msg_gemm_kernel<<<msgBlocks, TB, 0, stream>>>(x16, Wt1, sortedE, bstart, ei,
                                                inv, hbuf);
  relu_cvt_kernel<<<(NELT + TB - 1) / TB, TB, 0, stream>>>(hbuf, h16, NELT);

  // --- layer 2 ---
  dense_rowgemm_kernel<<<denseBlocks, TB, 0, stream>>>(h16, rT2, bias2, out,
                                                       NN);
  msg_gemm_kernel<<<msgBlocks, TB, 0, stream>>>(h16, Wt2, sortedE, bstart, ei,
                                                inv, out);
}
